// myBadTransfomer_1795296329851
// MI455X (gfx1250) — matrix operand order repaired and checked by reading, not compiled, not run on hardware
//
#include <hip/hip_runtime.h>
#include <hip/hip_bf16.h>

// ---------------------------------------------------------------------------
// myBadTransformer forward for MI455X (gfx1250), bf16 WMMA tensor-core path.
//   D=4096, B=2, S=2048.  All 6 GEMMs use v_wmma_f32_16x16x32_bf16 with fp32
//   accumulation, async global->LDS tile DMA (ASYNCcnt), and a 3-buffer LDS
//   pipeline with prefetch distance 2 to cover HBM latency.
//   GEMM form everywhere: C[m,n] = sum_k A[m,k] * B[n,k]
//   (A: MxK row-major, B: NxK row-major -> both operands K-contiguous).
// ---------------------------------------------------------------------------

typedef __attribute__((ext_vector_type(16))) __bf16 v16bf;
typedef __attribute__((ext_vector_type(8)))  float  v8f;
typedef __attribute__((ext_vector_type(4)))  unsigned int u32x4;

union Frag {
    u32x4 u[2];
    v16bf v;
};

__device__ __forceinline__ unsigned short f2bf_rne(float f) {
    unsigned int u = __builtin_bit_cast(unsigned int, f);
    u += 0x7FFFu + ((u >> 16) & 1u);
    return (unsigned short)(u >> 16);
}

// Low 32 bits of a generic pointer to __shared__ == LDS byte offset
// (flat->LDS mapping truncates the aperture bits, ISA 10.2).
__device__ __forceinline__ unsigned lds_off_u32(const void* p) {
    return (unsigned)(unsigned long long)p;
}

// gfx1250 async DMA: global -> LDS, 16B per lane, tracked by ASYNCcnt.
// GVS addressing: saddr(SGPR pair) + per-lane unsigned byte offset VGPR.
__device__ __forceinline__ void async_ld_b128(unsigned lds_off, const void* sbase,
                                              unsigned voff) {
    asm volatile("global_load_async_to_lds_b128 %0, %1, %2"
                 :: "v"(lds_off), "v"(voff), "s"(sbase)
                 : "memory");
}
// Wait until at most the youngest batch (6 asyncs) is still in flight:
// async loads complete in order, so this proves the older batch has landed.
__device__ __forceinline__ void wait_async_le6() {
    asm volatile("s_wait_asynccnt 6" ::: "memory");
}
__device__ __forceinline__ void wait_async0() {
    asm volatile("s_wait_asynccnt 0" ::: "memory");
}

// --------------------------- fp32 -> bf16 convert ---------------------------
__global__ __launch_bounds__(256)
void cvt_f32_bf16(const float* __restrict__ in, unsigned short* __restrict__ out, int n4) {
    int i = blockIdx.x * blockDim.x + threadIdx.x;
    int stride = gridDim.x * blockDim.x;
    for (; i < n4; i += stride) {
        float4 f = ((const float4*)in)[i];
        ushort2 lo; lo.x = f2bf_rne(f.x); lo.y = f2bf_rne(f.y);
        ushort2 hi; hi.x = f2bf_rne(f.z); hi.y = f2bf_rne(f.w);
        ((ushort2*)out)[2 * i]     = lo;
        ((ushort2*)out)[2 * i + 1] = hi;
    }
}

// ------------------------------ WMMA GEMM ----------------------------------
// Block: 256 threads (8 wave32), tile 128(M) x 256(N), BK=32.
// Wave grid 2(M) x 4(N); each wave computes 64x64 via 4x4 WMMA 16x16 tiles.
// EPI: 0 = store bf16
//      1 = store f32
//      2 = h = resid + silu(acc + bias[n])  -> bf16
//      3 = acc + bias[n]                    -> f32
#define BM 128
#define BN 256
#define BK 32
#define LDSS 40                     // padded LDS row stride in halves (32+8)
#define A_HALVES (BM * LDSS)        // 5120
#define B_HALVES (BN * LDSS)        // 10240
#define BUF_HALVES (A_HALVES + B_HALVES)   // 15360 halves = 30720 B / buffer

template <int EPI>
__global__ __launch_bounds__(256)
void gemm_bf16_wmma(const unsigned short* __restrict__ Abase,
                    const unsigned short* __restrict__ Bbase,
                    void* __restrict__ Cbase,
                    int M, int N, int K,
                    int lda, int ldb, int ldc,
                    long long strideA, long long strideB, long long strideC,
                    const float* __restrict__ bias,
                    const float* __restrict__ resid)
{
    (void)M; (void)N;  // all shapes here are multiples of the block tile

    __shared__ unsigned short lds[3 * BUF_HALVES];   // 92 KB, triple buffered

    const int tid  = threadIdx.x;
    const int lane = tid & 31;
    const int wave = tid >> 5;
    const int wm   = wave >> 2;   // 0..1 : wave row (2 waves over M, 64 each)
    const int wn   = wave & 3;    // 0..3 : wave col (4 waves over N, 64 each)
    const int z    = blockIdx.z;

    const unsigned short* A  = Abase + (size_t)z * (size_t)strideA;
    const unsigned short* Bm = Bbase + (size_t)z * (size_t)strideB;

    const int m0 = blockIdx.y * BM;
    const int n0 = blockIdx.x * BN;

    // DMA mapping: 4 threads per row * 8 halves (16B); 64 rows per pass.
    // A: 2 passes (128 rows), B: 4 passes (256 rows) -> 6 asyncs/thread/tile.
    const int lrow = tid >> 2;
    const int lcol = (tid & 3) * 8;

    unsigned aoff[2], boff[4], lA[2], lB[4];
#pragma unroll
    for (int p = 0; p < 2; ++p) {
        aoff[p] = (unsigned)(((m0 + lrow + p * 64) * (size_t)lda + lcol) * 2);
        lA[p]   = (unsigned)(((lrow + p * 64) * LDSS + lcol) * 2);
    }
#pragma unroll
    for (int q = 0; q < 4; ++q) {
        boff[q] = (unsigned)(((n0 + lrow + q * 64) * (size_t)ldb + lcol) * 2);
        lB[q]   = (unsigned)((A_HALVES + (lrow + q * 64) * LDSS + lcol) * 2);
    }
    const unsigned ldsbase = lds_off_u32(&lds[0]);

    const int half = lane >> 4;
    const int l16  = lane & 15;

    v8f acc[4][4];
#pragma unroll
    for (int i = 0; i < 4; ++i)
#pragma unroll
        for (int j = 0; j < 4; ++j) acc[i][j] = (v8f)0.0f;

    // prologue: DMA tiles 0 and 1 into buffers 0 and 1
    {
#pragma unroll
        for (int p = 0; p < 2; ++p) async_ld_b128(ldsbase + lA[p], A, aoff[p]);
#pragma unroll
        for (int q = 0; q < 4; ++q) async_ld_b128(ldsbase + lB[q], Bm, boff[q]);
        if (BK < K) {
            unsigned nb = ldsbase + BUF_HALVES * 2;
            unsigned ko = BK * 2;
#pragma unroll
            for (int p = 0; p < 2; ++p) async_ld_b128(nb + lA[p], A, aoff[p] + ko);
#pragma unroll
            for (int q = 0; q < 4; ++q) async_ld_b128(nb + lB[q], Bm, boff[q] + ko);
        }
    }

    unsigned b0 = 0, b1 = BUF_HALVES, b2 = 2 * BUF_HALVES;  // half offsets
    for (int kk = 0; kk < K; kk += BK) {
        // Tile kk landed iff at most the (optional) tile kk+BK batch remains
        // in flight; async loads complete in order.
        if (kk + BK < K) wait_async_le6(); else wait_async0();
        __syncthreads();   // everyone's tile-kk writes landed; reads of b2's
                           // previous contents (last iteration's b0) drained at this barrier

        // prefetch tile kk+2*BK into b2 (overlaps frag loads + 16 WMMAs)
        if (kk + 2 * BK < K) {
            unsigned nb = ldsbase + b2 * 2;
            unsigned ko = (unsigned)((kk + 2 * BK) * 2);
#pragma unroll
            for (int p = 0; p < 2; ++p) async_ld_b128(nb + lA[p], A, aoff[p] + ko);
#pragma unroll
            for (int q = 0; q < 4; ++q) async_ld_b128(nb + lB[q], Bm, boff[q] + ko);
        }

        const unsigned short* buf = lds + b0;

        // A fragment (16x32 bf16, ISA 7.12.2): lane m=l16; elems 0-7 at
        // K=half*8.., elems 8-15 at K=16+half*8..
        Frag a[4], b[4];
#pragma unroll
        for (int i = 0; i < 4; ++i) {
            const unsigned short* p = &buf[(wm * 64 + i * 16 + l16) * LDSS + half * 8];
            a[i].u[0] = *(const u32x4*)p;
            a[i].u[1] = *(const u32x4*)(p + 16);
        }
        // B fragment (32x16 bf16): lane n=l16; elems 0-7 at K=half*8..,
        // elems 8-15 at K=16+half*8..
#pragma unroll
        for (int j = 0; j < 4; ++j) {
            const unsigned short* p =
                &buf[A_HALVES + (wn * 64 + j * 16 + l16) * LDSS + half * 8];
            b[j].u[0] = *(const u32x4*)p;
            b[j].u[1] = *(const u32x4*)(p + 16);
        }

#pragma unroll
        for (int i = 0; i < 4; ++i)
#pragma unroll
            for (int j = 0; j < 4; ++j)
                acc[i][j] = __builtin_amdgcn_wmma_f32_16x16x32_bf16(
                    false, a[i].v, false, b[j].v, (short)0, acc[i][j], false, false);

        unsigned t = b0; b0 = b1; b1 = b2; b2 = t;   // rotate ring buffers
    }

    // Epilogue.  C layout: VGPR r -> M = r + 8*half, lane -> N = l16.
#pragma unroll
    for (int i = 0; i < 4; ++i) {
#pragma unroll
        for (int j = 0; j < 4; ++j) {
            int mg0 = m0 + wm * 64 + i * 16 + half * 8;
            int ng  = n0 + wn * 64 + j * 16 + l16;
#pragma unroll
            for (int r = 0; r < 8; ++r) {
                int mg = mg0 + r;
                float v = acc[i][j][r];
                size_t idx = (size_t)mg * ldc + ng;
                if (EPI == 0) {
                    ((unsigned short*)Cbase)[(size_t)z * (size_t)strideC + idx] = f2bf_rne(v);
                } else if (EPI == 1) {
                    ((float*)Cbase)[(size_t)z * (size_t)strideC + idx] = v;
                } else if (EPI == 2) {
                    float t = v + bias[ng];
                    float s = t * (1.0f / (1.0f + __expf(-t)));   // silu
                    ((unsigned short*)Cbase)[idx] = f2bf_rne(resid[idx] + s);
                } else {
                    ((float*)Cbase)[idx] = v + bias[ng];
                }
            }
        }
    }
}

// --------------------------- softmax row statistics -------------------------
__global__ __launch_bounds__(256)
void softmax_rowstats(const float* __restrict__ scores,
                      float* __restrict__ rmax, float* __restrict__ rinv, int Dn) {
    __shared__ float sd[256];
    const float* row = scores + (size_t)blockIdx.x * Dn;
    int tid = threadIdx.x;

    float m = -3.402823466e38f;
    for (int i = tid; i < Dn; i += 256) m = fmaxf(m, row[i]);
    sd[tid] = m; __syncthreads();
    for (int s = 128; s > 0; s >>= 1) {
        if (tid < s) sd[tid] = fmaxf(sd[tid], sd[tid + s]);
        __syncthreads();
    }
    m = sd[0]; __syncthreads();

    float sum = 0.0f;
    for (int i = tid; i < Dn; i += 256) sum += __expf(row[i] - m);
    sd[tid] = sum; __syncthreads();
    for (int s = 128; s > 0; s >>= 1) {
        if (tid < s) sd[tid] += sd[tid + s];
        __syncthreads();
    }
    if (tid == 0) { rmax[blockIdx.x] = m; rinv[blockIdx.x] = 1.0f / sd[0]; }
}

// ------------------- softmax normalize + transpose -> bf16 ------------------
// attnT[b][e][d] = softmax(scores[b][d][:])[e], 32x32 LDS tiles, coalesced.
__global__ __launch_bounds__(1024)
void softmax_transpose(const float* __restrict__ scores,
                       const float* __restrict__ rmax,
                       const float* __restrict__ rinv,
                       unsigned short* __restrict__ attnT, int Dn) {
    __shared__ unsigned short tile[32][33];
    int b  = blockIdx.z;
    int tx = threadIdx.x, ty = threadIdx.y;
    size_t base = (size_t)b * Dn * Dn;

    int d = blockIdx.y * 32 + ty;
    int e = blockIdx.x * 32 + tx;
    int rowid = b * Dn + d;
    float v = scores[base + (size_t)d * Dn + e];
    float p = __expf(v - rmax[rowid]) * rinv[rowid];
    tile[ty][tx] = f2bf_rne(p);
    __syncthreads();

    int eo = blockIdx.x * 32 + ty;
    int d2 = blockIdx.y * 32 + tx;
    attnT[base + (size_t)eo * Dn + d2] = tile[tx][ty];
}

// ------------------------------- launcher -----------------------------------
extern "C" void kernel_launch(void* const* d_in, const int* in_sizes, int n_in,
                              void* d_out, int out_size, void* d_ws, size_t ws_size,
                              hipStream_t stream) {
    (void)in_sizes; (void)n_in; (void)out_size; (void)ws_size;

    const float* x    = (const float*)d_in[0];
    const float* W1   = (const float*)d_in[1];
    const float* W2   = (const float*)d_in[2];
    const float* W4   = (const float*)d_in[3];
    const float* b4   = (const float*)d_in[4];
    const float* Wout = (const float*)d_in[5];
    const float* bout = (const float*)d_in[6];
    float* out = (float*)d_out;

    const int Dn = 4096, Bb = 2, S = 2048, BS = Bb * S;   // BS = 4096
    const size_t NE = (size_t)Dn * Dn;                    // 16.78M (= BS*Dn too)
    const size_t MB = 1ull << 20;

    char* w = (char*)d_ws;
    unsigned short* xbf    = (unsigned short*)(w + 0);        // 32 MB
    unsigned short* W1bf   = (unsigned short*)(w + 32 * MB);
    unsigned short* W2bf   = (unsigned short*)(w + 64 * MB);
    unsigned short* W4bf   = (unsigned short*)(w + 96 * MB);
    unsigned short* Woutbf = (unsigned short*)(w + 128 * MB);
    unsigned short* xAT    = (unsigned short*)(w + 160 * MB); // [D][BS] bf16, 32 MB
    unsigned short* xBT    = (unsigned short*)(w + 192 * MB); // [D][BS] bf16, 32 MB
    float*          scores = (float*)(w + 224 * MB);          // [B][D][D] f32, 128 MB
    float*          rmax   = (float*)(w + 352 * MB);          // B*D f32
    float*          rinv   = (float*)(w + 352 * MB + 64 * 1024);
    // dead-buffer reuse:
    unsigned short* attnT  = (unsigned short*)(w + 160 * MB); // [B][D][D] bf16, 64 MB (over xAT+xBT)
    unsigned short* xO     = (unsigned short*)(w + 224 * MB); // [B][S][D] bf16, 32 MB (over scores)
    unsigned short* hbuf   = (unsigned short*)(w + 256 * MB); // [BS][D] bf16, 32 MB

    // 1) bf16 conversions of x and weights
    cvt_f32_bf16<<<2048, 256, 0, stream>>>(x,    xbf,    (int)(NE / 4));
    cvt_f32_bf16<<<2048, 256, 0, stream>>>(W1,   W1bf,   (int)(NE / 4));
    cvt_f32_bf16<<<2048, 256, 0, stream>>>(W2,   W2bf,   (int)(NE / 4));
    cvt_f32_bf16<<<2048, 256, 0, stream>>>(W4,   W4bf,   (int)(NE / 4));
    cvt_f32_bf16<<<2048, 256, 0, stream>>>(Wout, Woutbf, (int)(NE / 4));

    // 2) xAT[d][s_glob] = sum_k W1[d][k] x[s_glob][k]   (== xA transposed)
    gemm_bf16_wmma<0><<<dim3(BS / BN, Dn / BM, 1), 256, 0, stream>>>(
        W1bf, xbf, xAT, Dn, BS, Dn, Dn, Dn, BS, 0, 0, 0, nullptr, nullptr);
    gemm_bf16_wmma<0><<<dim3(BS / BN, Dn / BM, 1), 256, 0, stream>>>(
        W2bf, xbf, xBT, Dn, BS, Dn, Dn, Dn, BS, 0, 0, 0, nullptr, nullptr);

    // 3) scores[b][d][e] = sum_s xAT[d][b*S+s] * xBT[e][b*S+s]
    gemm_bf16_wmma<1><<<dim3(Dn / BN, Dn / BM, Bb), 256, 0, stream>>>(
        xAT, xBT, scores, Dn, Dn, S, BS, BS, Dn,
        (long long)S, (long long)S, (long long)Dn * Dn, nullptr, nullptr);

    // 4) softmax over e, emit attnT bf16
    softmax_rowstats<<<Bb * Dn, 256, 0, stream>>>(scores, rmax, rinv, Dn);
    softmax_transpose<<<dim3(Dn / 32, Dn / 32, Bb), dim3(32, 32), 0, stream>>>(
        scores, rmax, rinv, attnT, Dn);

    // 5) xO[b][s][e] = sum_d x[b][s][d] * attnT[b][e][d]
    gemm_bf16_wmma<0><<<dim3(Dn / BN, S / BM, Bb), 256, 0, stream>>>(
        xbf, attnT, xO, S, Dn, Dn, Dn, Dn, Dn,
        (long long)S * Dn, (long long)Dn * Dn, (long long)S * Dn, nullptr, nullptr);

    // 6) h = x + silu(xO @ W4^T + b4)   -> bf16
    gemm_bf16_wmma<2><<<dim3(Dn / BN, BS / BM, 1), 256, 0, stream>>>(
        xO, W4bf, hbuf, BS, Dn, Dn, Dn, Dn, Dn, 0, 0, 0, b4, x);

    // 7) out = h @ Wout^T + bout        -> f32
    gemm_bf16_wmma<3><<<dim3(Dn / BN, BS / BM, 1), 256, 0, stream>>>(
        hbuf, Woutbf, out, BS, Dn, Dn, Dn, Dn, Dn, 0, 0, 0, bout, nullptr);
}
